// GDNEncoderBlock_7103875907806
// MI455X (gfx1250) — matrix operand order repaired and checked by reading, not compiled, not run on hardware
//
#include <hip/hip_runtime.h>
#include <hip/hip_bf16.h>
#include <math.h>
#include <stdint.h>

// ---------------------------------------------------------------------------
// GDN encoder block for MI455X (gfx1250, wave32, WMMA + Tensor Data Mover)
//   - GEMMs: bf16 v_wmma_f32_16x16x32_bf16, 64x64 tile / WG, K staged 64 deep
//     via TDM (tensor_load_to_lds) double-buffered in LDS with HW padding.
//   - Scan: rank-1 delta-rule updates, LDS-resident state, ds_add_f32.
// ---------------------------------------------------------------------------

typedef __bf16 bf16_t;
typedef __attribute__((ext_vector_type(8)))  bf16_t v8bf;
typedef __attribute__((ext_vector_type(16))) bf16_t v16bf;
typedef __attribute__((ext_vector_type(8)))  float  v8f;
typedef __attribute__((ext_vector_type(4)))  unsigned int v4u;
typedef __attribute__((ext_vector_type(8)))  int v8i;
typedef __attribute__((ext_vector_type(4)))  int v4i;

#define BATCH   8
#define NSEQ    512
#define CDIM    512
#define NHEAD   8
#define KCONV   4
#define DHEAD   64
#define HIDDEN  2048
#define MROWS   (BATCH * NSEQ)   // 4096

// -------------------------------- helpers ----------------------------------

__device__ __forceinline__ float act_silu(float x) { return x / (1.0f + __expf(-x)); }
__device__ __forceinline__ float act_sigm(float x) { return 1.0f / (1.0f + __expf(-x)); }
__device__ __forceinline__ float act_gelu(float x) { return 0.5f * x * (1.0f + erff(x * 0.70710678118654752f)); }

__device__ __forceinline__ float block_reduce_sum(float v, float* red, int t) {
    red[t] = v;
    __syncthreads();
#pragma unroll
    for (int s = 128; s > 0; s >>= 1) {
        if (t < s) red[t] += red[t + s];
        __syncthreads();
    }
    float r = red[0];
    __syncthreads();
    return r;
}

// --------------------------- TDM tile load ---------------------------------
// DMA one 64-row x 64-element bf16 tile (row stride Kd elements in memory)
// into LDS at lds_off, padding each 128B row with 16B -> LDS row stride 144B
// (= 72 bf16), matching the As/Bs[64][72] layout.
//
// D# per CDNA5 ISA:
//   group0: [1:0]=count=1 | lds_addr | global_addr[56:0] | type=2
//   group1: data_size=1(2B), pad_enable, pad_interval=4 (32 dw), pad_amount=3
//           (4 dw), tensor_dim0=Kd, tensor_dim1=big, tile_dim0=64, tile_dim1=64,
//           tensor_dim0_stride=Kd
//   remaining groups: zero (2-D tensor)
// Toolchain here is the 6-arg clang-23 form:
//   (uint32x4 g0, int32x8 g1, int32x4, int32x4, int32x8, i32 cpol)

__device__ __forceinline__ void tdm_load_tile_2d(unsigned lds_off,
                                                 const bf16_t* gaddr, int Kd) {
    const unsigned long long ga = (unsigned long long)(uintptr_t)gaddr;
    v4u g0;
    g0[0] = 1u;                                    // count=1 (valid), user desc
    g0[1] = lds_off;                               // LDS byte address
    g0[2] = (unsigned)ga;                          // global_addr[31:0]
    g0[3] = (unsigned)((ga >> 32) & 0x01FFFFFFu)   // global_addr[56:32]
          | (2u << 30);                            // type = 2 ("image")
    const unsigned w0 = (1u << 16)                 // data_size = 2 bytes
                      | (1u << 20)                 // pad_enable
                      | (4u << 22)                 // pad_interval: 32 DWORDs
                      | (3u << 25);                // pad_amount : 4 DWORDs
    const unsigned d0 = (unsigned)Kd;              // tensor_dim0 (elements)
    const unsigned d1 = 1u << 20;                  // tensor_dim1 (rows, generous)
    v8i g1;
    g1[0] = (int)w0;
    g1[1] = (int)((d0 & 0xFFFFu) << 16);                       // dim0 lo16
    g1[2] = (int)((d0 >> 16) | ((d1 & 0xFFFFu) << 16));        // dim0 hi|dim1 lo
    g1[3] = (int)((d1 >> 16) | (64u << 16));                   // dim1 hi|tile_dim0
    g1[4] = (int)64u;                                          // tile_dim1 (tile_dim2=0)
    g1[5] = (int)(unsigned)Kd;                                 // dim0_stride lo32
    g1[6] = 0;                                                 // dim0_stride hi16
    g1[7] = 0;
    const v4i gz4 = {0, 0, 0, 0};
    const v8i gz8 = {0, 0, 0, 0, 0, 0, 0, 0};
    __builtin_amdgcn_tensor_load_to_lds(g0, g1, gz4, gz4, gz8, 0);
}

// ------------------------------ convert ------------------------------------

__global__ __launch_bounds__(256) void f32_to_bf16_kernel(const float* __restrict__ in,
                                                          bf16_t* __restrict__ out, int n) {
    int i = blockIdx.x * 256 + threadIdx.x;
    if (i < n) out[i] = (bf16_t)in[i];
}

// ------------------------------ layernorm ----------------------------------

__global__ __launch_bounds__(256) void layernorm_to_bf16(const float* __restrict__ x,
                                                         const float* __restrict__ g,
                                                         const float* __restrict__ b,
                                                         bf16_t* __restrict__ out) {
    __shared__ float red[256];
    const int row = blockIdx.x;
    const int t   = threadIdx.x;
    const float* xr = x + (size_t)row * CDIM;
    float v0 = xr[t], v1 = xr[t + 256];
    float mean = block_reduce_sum(v0 + v1, red, t) * (1.0f / CDIM);
    float d0 = v0 - mean, d1 = v1 - mean;
    float var = block_reduce_sum(d0 * d0 + d1 * d1, red, t) * (1.0f / CDIM);
    float rstd = rsqrtf(var + 1e-5f);
    bf16_t* orow = out + (size_t)row * CDIM;
    orow[t]       = (bf16_t)(d0 * rstd * g[t]       + b[t]);
    orow[t + 256] = (bf16_t)(d1 * rstd * g[t + 256] + b[t + 256]);
}

// --------------------------- WMMA bf16 GEMM --------------------------------
// out[M,Nc] = ACT(A[M,Kd] * W[Nc,Kd]^T + bias) (+ resid).
// 64x64 tile / 256-thread WG (8 waves). Wave (wm,wn) owns a 16x32 sub-tile.
// K staged 64 deep via TDM into double-buffered padded LDS; 4 WMMAs/stage.

template <int ACT>  // 0=none, 1=silu, 2=gelu
__global__ __launch_bounds__(256) void gemm_bf16_wmma(const bf16_t* __restrict__ A,
                                                      const bf16_t* __restrict__ W,
                                                      const float*  __restrict__ bias,
                                                      const float*  __restrict__ resid,
                                                      float*        __restrict__ outF,
                                                      bf16_t*       __restrict__ outB,
                                                      int Nc, int Kd) {
    __shared__ bf16_t As[2][64][72];   // row stride 144B (TDM pad), 16B aligned
    __shared__ bf16_t Bs[2][64][72];

    const int t    = threadIdx.x;
    const int lane = t & 31;
    const int wave = t >> 5;
    const int wm   = wave & 3;        // 4 row groups of 16
    const int wn   = wave >> 2;       // 2 col groups of 32
    const int half = lane >> 4;       // lane half (0/1)
    const int l16  = lane & 15;

    const int rowBase = blockIdx.y * 64;
    const int colBase = blockIdx.x * 64;

    const bf16_t* aTile = A + (size_t)rowBase * Kd;
    const bf16_t* wTile = W + (size_t)colBase * Kd;

    v8f acc0 = {};
    v8f acc1 = {};

    // preload K-tile 0 (one wave issues the DMA; TENSORcnt is per-wave)
    if (t < 32) {
        tdm_load_tile_2d((unsigned)(uintptr_t)&As[0][0][0], aTile, Kd);
        tdm_load_tile_2d((unsigned)(uintptr_t)&Bs[0][0][0], wTile, Kd);
    }

    int buf = 0;
    for (int k0 = 0; k0 < Kd; k0 += 64) {
        if (t < 32) __builtin_amdgcn_s_wait_tensorcnt(0);
        __syncthreads();   // tile `buf` visible to all waves

        if ((k0 + 64) < Kd && t < 32) {   // kick off next tile into other buffer
            tdm_load_tile_2d((unsigned)(uintptr_t)&As[buf ^ 1][0][0], aTile + k0 + 64, Kd);
            tdm_load_tile_2d((unsigned)(uintptr_t)&Bs[buf ^ 1][0][0], wTile + k0 + 64, Kd);
        }

#pragma unroll
        for (int kk = 0; kk < 2; ++kk) {
            const int kb = kk * 32;
            // A fragment (16x32): lane<16 K={0..7,16..23}, lane>=16 K={8..15,24..31}
            v8bf alo = *(const v8bf*)&As[buf][wm * 16 + l16][kb + half * 8];
            v8bf ahi = *(const v8bf*)&As[buf][wm * 16 + l16][kb + 16 + half * 8];
            v16bf af;
#pragma unroll
            for (int i = 0; i < 8; ++i) { af[i] = alo[i]; af[8 + i] = ahi[i]; }

            // B fragments (32x16): lane<16 K={0..7,16..23} of col l16, lane>=16 K={8..15,24..31}
            v8bf b0l = *(const v8bf*)&Bs[buf][wn * 32 + l16][kb + half * 8];
            v8bf b0h = *(const v8bf*)&Bs[buf][wn * 32 + l16][kb + 16 + half * 8];
            v8bf b1l = *(const v8bf*)&Bs[buf][wn * 32 + 16 + l16][kb + half * 8];
            v8bf b1h = *(const v8bf*)&Bs[buf][wn * 32 + 16 + l16][kb + 16 + half * 8];
            v16bf bf0, bf1;
#pragma unroll
            for (int i = 0; i < 8; ++i) {
                bf0[i] = b0l[i]; bf0[8 + i] = b0h[i];
                bf1[i] = b1l[i]; bf1[8 + i] = b1h[i];
            }

            acc0 = __builtin_amdgcn_wmma_f32_16x16x32_bf16(false, af, false, bf0,
                                                           (short)0, acc0, false, false);
            acc1 = __builtin_amdgcn_wmma_f32_16x16x32_bf16(false, af, false, bf1,
                                                           (short)0, acc1, false, false);
        }
        buf ^= 1;
    }

    // C/D layout: lane<16 -> M = wm*16 + r ; lane>=16 -> M = wm*16 + 8 + r
    const int orow0 = rowBase + wm * 16 + half * 8;
#pragma unroll
    for (int n = 0; n < 2; ++n) {
        const v8f& acc = n ? acc1 : acc0;
        const int col = colBase + wn * 32 + n * 16 + l16;
        const float bv = bias ? bias[col] : 0.0f;
#pragma unroll
        for (int r = 0; r < 8; ++r) {
            float v = acc[r] + bv;
            if (ACT == 1)      v = act_silu(v);
            else if (ACT == 2) v = act_gelu(v);
            const size_t idx = (size_t)(orow0 + r) * Nc + col;
            if (resid) v += resid[idx];
            if (outF) outF[idx] = v;
            if (outB) outB[idx] = (bf16_t)v;
        }
    }
}

// ------------------------------- gates -------------------------------------

__global__ __launch_bounds__(256) void gates_kernel(const bf16_t* __restrict__ h,
                                                    const float* __restrict__ wa,
                                                    const float* __restrict__ ba,
                                                    const float* __restrict__ wb,
                                                    const float* __restrict__ bb,
                                                    float* __restrict__ ga,
                                                    float* __restrict__ gb) {
    __shared__ float red[256];
    const int row = blockIdx.x;
    const int t   = threadIdx.x;
    const float h0 = (float)h[(size_t)row * CDIM + t];
    const float h1 = (float)h[(size_t)row * CDIM + t + 256];
    for (int hh = 0; hh < NHEAD; ++hh) {
        float da = block_reduce_sum(h0 * wa[hh * CDIM + t] + h1 * wa[hh * CDIM + t + 256], red, t);
        if (t == 0) ga[(size_t)row * NHEAD + hh] = act_sigm(da + ba[hh]);
        float db = block_reduce_sum(h0 * wb[hh * CDIM + t] + h1 * wb[hh * CDIM + t + 256], red, t);
        if (t == 0) gb[(size_t)row * NHEAD + hh] = act_sigm(db + bb[hh]);
    }
}

// -------------------- depthwise conv + silu (+ L2 norm) --------------------

__device__ __forceinline__ float dw_one(const float* __restrict__ base, int b, int n, int c,
                                        const float* __restrict__ w) {
    float s = 0.0f;
#pragma unroll
    for (int j = 0; j < KCONV; ++j) {
        const int idx = n - (KCONV / 2) + j;
        if (idx >= 0 && idx < NSEQ)
            s += base[((size_t)b * NSEQ + idx) * CDIM + c] * w[c * KCONV + j];
    }
    return s;
}

__global__ __launch_bounds__(256) void dwconv_kernel(const float* __restrict__ qp,
                                                     const float* __restrict__ kp,
                                                     const float* __restrict__ vp,
                                                     const float* __restrict__ cq,
                                                     const float* __restrict__ ck,
                                                     const float* __restrict__ cv,
                                                     float* __restrict__ qo,
                                                     float* __restrict__ ko,
                                                     float* __restrict__ vo) {
    __shared__ float red[256];
    const int row = blockIdx.x;
    const int b   = row / NSEQ;
    const int n   = row % NSEQ;
    const int t   = threadIdx.x;

    float q0 = act_silu(dw_one(qp, b, n, t, cq));
    float q1 = act_silu(dw_one(qp, b, n, t + 256, cq));
    float qs = block_reduce_sum(q0 * q0 + q1 * q1, red, t);
    float qr = 1.0f / (sqrtf(qs) + 1e-6f);
    qo[(size_t)row * CDIM + t]       = q0 * qr;
    qo[(size_t)row * CDIM + t + 256] = q1 * qr;

    float k0 = act_silu(dw_one(kp, b, n, t, ck));
    float k1 = act_silu(dw_one(kp, b, n, t + 256, ck));
    float ks = block_reduce_sum(k0 * k0 + k1 * k1, red, t);
    float kr = 1.0f / (sqrtf(ks) + 1e-6f);
    ko[(size_t)row * CDIM + t]       = k0 * kr;
    ko[(size_t)row * CDIM + t + 256] = k1 * kr;

    vo[(size_t)row * CDIM + t]       = act_silu(dw_one(vp, b, n, t, cv));
    vo[(size_t)row * CDIM + t + 256] = act_silu(dw_one(vp, b, n, t + 256, cv));
}

// ------------------------- gated delta-rule scan ---------------------------

__global__ __launch_bounds__(256) void gdn_scan_kernel(const float* __restrict__ qn,
                                                       const float* __restrict__ kn,
                                                       const float* __restrict__ vn,
                                                       const float* __restrict__ ga,
                                                       const float* __restrict__ gb,
                                                       bf16_t* __restrict__ O) {
    __shared__ float S[DHEAD][DHEAD + 1];
    __shared__ float kv[3 * DHEAD];  // k | v | q
    __shared__ float oacc[DHEAD];
    __shared__ float ab2[2];

    const int bh = blockIdx.x;
    const int b  = bh / NHEAD;
    const int h  = bh % NHEAD;
    const int t  = threadIdx.x;
    const int d  = t >> 2;
    const int f0 = (t & 3) << 4;

#pragma unroll
    for (int i = 0; i < 16; ++i) S[d][f0 + i] = 0.0f;

    for (int n = 0; n < NSEQ; ++n) {
        const size_t row  = (size_t)b * NSEQ + n;
        const size_t base = row * CDIM + h * DHEAD;
        __syncthreads();
        if (t < 64)        kv[t]  = kn[base + t];
        else if (t < 128)  kv[t]  = vn[base + (t - 64)];
        else if (t < 192)  kv[t]  = qn[base + (t - 128)];
        else if (t == 192) ab2[0] = ga[row * NHEAD + h];
        else if (t == 193) ab2[1] = gb[row * NHEAD + h];
        if (t < 64) oacc[t] = 0.0f;
        __syncthreads();

        const float a_ = ab2[0];
        const float b_ = ab2[1];

        float p = 0.0f;
#pragma unroll
        for (int i = 0; i < 16; ++i) p += S[d][f0 + i] * kv[f0 + i];
        p += __shfl_xor(p, 1, 32);
        p += __shfl_xor(p, 2, 32);
        const float Sk = p;
        const float vd = kv[DHEAD + d];
        const float qd = kv[2 * DHEAD + d];

#pragma unroll
        for (int i = 0; i < 16; ++i) {
            const float kf = kv[f0 + i];
            const float s  = a_ * S[d][f0 + i] + (b_ * vd - a_ * b_ * Sk) * kf;
            S[d][f0 + i] = s;
            atomicAdd(&oacc[f0 + i], qd * s);   // ds_add_f32
        }
        __syncthreads();
        if (t < 64) O[base + t] = (bf16_t)oacc[t];
    }
}

// ------------------------------ launcher -----------------------------------

extern "C" void kernel_launch(void* const* d_in, const int* in_sizes, int n_in,
                              void* d_out, int out_size, void* d_ws, size_t ws_size,
                              hipStream_t stream) {
    const float* x    = (const float*)d_in[0];
    const float* ln1g = (const float*)d_in[1];
    const float* ln1b = (const float*)d_in[2];
    const float* wq   = (const float*)d_in[3];
    const float* bq   = (const float*)d_in[4];
    const float* wk   = (const float*)d_in[5];
    const float* bk   = (const float*)d_in[6];
    const float* wv   = (const float*)d_in[7];
    const float* bv   = (const float*)d_in[8];
    const float* wa   = (const float*)d_in[9];
    const float* ba   = (const float*)d_in[10];
    const float* wb   = (const float*)d_in[11];
    const float* bb   = (const float*)d_in[12];
    const float* cq   = (const float*)d_in[13];
    const float* ck   = (const float*)d_in[14];
    const float* cv   = (const float*)d_in[15];
    const float* wo   = (const float*)d_in[16];
    const float* bo   = (const float*)d_in[17];
    const float* ln2g = (const float*)d_in[18];
    const float* ln2b = (const float*)d_in[19];
    const float* w1   = (const float*)d_in[20];
    const float* b1   = (const float*)d_in[21];
    const float* w2   = (const float*)d_in[22];
    const float* b2   = (const float*)d_in[23];
    float* out = (float*)d_out;

    char*  ws  = (char*)d_ws;
    size_t off = 0;
    auto carve = [&](size_t bytes) -> char* {
        off = (off + 255) & ~(size_t)255;
        char* p = ws + off;
        off += bytes;
        return p;
    };

    bf16_t* h_bf  = (bf16_t*)carve((size_t)MROWS * CDIM * sizeof(bf16_t));
    bf16_t* wq_bf = (bf16_t*)carve((size_t)CDIM * CDIM * sizeof(bf16_t));
    bf16_t* wk_bf = (bf16_t*)carve((size_t)CDIM * CDIM * sizeof(bf16_t));
    bf16_t* wv_bf = (bf16_t*)carve((size_t)CDIM * CDIM * sizeof(bf16_t));
    bf16_t* wo_bf = (bf16_t*)carve((size_t)CDIM * CDIM * sizeof(bf16_t));
    bf16_t* w1_bf = (bf16_t*)carve((size_t)HIDDEN * CDIM * sizeof(bf16_t));
    bf16_t* w2_bf = (bf16_t*)carve((size_t)CDIM * HIDDEN * sizeof(bf16_t));
    float*  qpre  = (float*)carve((size_t)MROWS * CDIM * sizeof(float));
    float*  kpre  = (float*)carve((size_t)MROWS * CDIM * sizeof(float));
    float*  vpre  = (float*)carve((size_t)MROWS * CDIM * sizeof(float));
    float*  qnb   = (float*)carve((size_t)MROWS * CDIM * sizeof(float));
    float*  knb   = (float*)carve((size_t)MROWS * CDIM * sizeof(float));
    float*  vnb   = (float*)carve((size_t)MROWS * CDIM * sizeof(float));
    float*  gaB   = (float*)carve((size_t)MROWS * NHEAD * sizeof(float));
    float*  gbB   = (float*)carve((size_t)MROWS * NHEAD * sizeof(float));
    bf16_t* O_bf  = (bf16_t*)carve((size_t)MROWS * CDIM * sizeof(bf16_t));
    float*  x1    = (float*)carve((size_t)MROWS * CDIM * sizeof(float));
    bf16_t* h2_bf = (bf16_t*)carve((size_t)MROWS * CDIM * sizeof(bf16_t));
    bf16_t* m_bf  = (bf16_t*)carve((size_t)MROWS * HIDDEN * sizeof(bf16_t));

    const int nw = CDIM * CDIM;
    f32_to_bf16_kernel<<<(nw + 255) / 256, 256, 0, stream>>>(wq, wq_bf, nw);
    f32_to_bf16_kernel<<<(nw + 255) / 256, 256, 0, stream>>>(wk, wk_bf, nw);
    f32_to_bf16_kernel<<<(nw + 255) / 256, 256, 0, stream>>>(wv, wv_bf, nw);
    f32_to_bf16_kernel<<<(nw + 255) / 256, 256, 0, stream>>>(wo, wo_bf, nw);
    const int nh = HIDDEN * CDIM;
    f32_to_bf16_kernel<<<(nh + 255) / 256, 256, 0, stream>>>(w1, w1_bf, nh);
    f32_to_bf16_kernel<<<(nh + 255) / 256, 256, 0, stream>>>(w2, w2_bf, nh);

    layernorm_to_bf16<<<MROWS, 256, 0, stream>>>(x, ln1g, ln1b, h_bf);

    dim3 g512(CDIM / 64, MROWS / 64);
    dim3 g2048(HIDDEN / 64, MROWS / 64);
    gemm_bf16_wmma<1><<<g512, 256, 0, stream>>>(h_bf, wq_bf, bq, nullptr, qpre, nullptr, CDIM, CDIM);
    gemm_bf16_wmma<1><<<g512, 256, 0, stream>>>(h_bf, wk_bf, bk, nullptr, kpre, nullptr, CDIM, CDIM);
    gemm_bf16_wmma<1><<<g512, 256, 0, stream>>>(h_bf, wv_bf, bv, nullptr, vpre, nullptr, CDIM, CDIM);

    gates_kernel<<<MROWS, 256, 0, stream>>>(h_bf, wa, ba, wb, bb, gaB, gbB);
    dwconv_kernel<<<MROWS, 256, 0, stream>>>(qpre, kpre, vpre, cq, ck, cv, qnb, knb, vnb);
    gdn_scan_kernel<<<BATCH * NHEAD, 256, 0, stream>>>(qnb, knb, vnb, gaB, gbB, O_bf);

    gemm_bf16_wmma<0><<<g512, 256, 0, stream>>>(O_bf, wo_bf, bo, x, x1, nullptr, CDIM, CDIM);
    layernorm_to_bf16<<<MROWS, 256, 0, stream>>>(x1, ln2g, ln2b, h2_bf);
    gemm_bf16_wmma<2><<<g2048, 256, 0, stream>>>(h2_bf, w1_bf, b1, nullptr, nullptr, m_bf, HIDDEN, CDIM);
    gemm_bf16_wmma<0><<<g512, 256, 0, stream>>>(m_bf, w2_bf, b2, x1, out, nullptr, CDIM, HIDDEN);
}
